// MultiHeadAttention_36532991819846
// MI455X (gfx1250) — matrix operand order repaired and checked by reading, not compiled, not run on hardware
//
#include <hip/hip_runtime.h>

// ---------------------------------------------------------------------------
// MultiHeadAttention on MI455X (gfx1250, wave32).
//   Stage 1: w_out fp32 -> bf16
//   Stage 2: neighbor-head sum + conv + tiny attention -> A matrix (bf16)
//   Stage 3: out = A @ w_out^T + b_out via v_wmma_f32_16x16x32_bf16, with
//            double-buffered GLOBAL_LOAD_ASYNC_TO_LDS (ASYNCcnt) tile fills.
// Workspace layout: [A_bf16: 7168*4096 u16][W_bf16: 4096*4096 u16] ~= 92.3 MB
// ---------------------------------------------------------------------------

typedef __attribute__((ext_vector_type(16))) __bf16 v16bf;
typedef __attribute__((ext_vector_type(8)))  __bf16 v8bf;
typedef __attribute__((ext_vector_type(8)))  float  v8f;
typedef __attribute__((ext_vector_type(4)))  int    v4i;

// address-space-qualified int4 pointee types for the async-LDS builtin
typedef v4i __attribute__((address_space(1))) gv4i;  // global
typedef v4i __attribute__((address_space(3))) lv4i;  // LDS

#if defined(__HIP_DEVICE_COMPILE__) &&                                     \
    __has_builtin(__builtin_amdgcn_global_load_async_to_lds_b128) &&       \
    __has_builtin(__builtin_amdgcn_s_wait_asynccnt)
#define USE_ASYNC_LDS 1
#else
#define USE_ASYNC_LDS 0
#endif

#if USE_ASYNC_LDS
// generic->AS(1): full 64-bit VA; generic->AS(3): low 32 bits are the LDS
// offset.
#define ASYNC_LOAD_B128(gsrc, ldst)                                         \
  __builtin_amdgcn_global_load_async_to_lds_b128(                           \
      (gv4i*)(unsigned long long)(const void*)(gsrc),                       \
      (lv4i*)(unsigned int)(unsigned long long)(const void*)(ldst),         \
      0, 0)
#endif

__device__ __forceinline__ unsigned short f2bf(float f) {
  // round-to-nearest-even fp32 -> bf16
  unsigned u = __float_as_uint(f);
  unsigned r = 0x7FFFu + ((u >> 16) & 1u);
  return (unsigned short)((u + r) >> 16);
}

// ---------------------------------------------------------------------------
// Stage 1: convert w_out to bf16
// ---------------------------------------------------------------------------
__global__ void cvt_w_kernel(const float* __restrict__ w,
                             unsigned short* __restrict__ wb, int n) {
  int i = blockIdx.x * blockDim.x + threadIdx.x;
  int stride = gridDim.x * blockDim.x;
  for (; i < n; i += stride) wb[i] = f2bf(w[i]);
}

// ---------------------------------------------------------------------------
// Stage 2: per (b,h): neighbor-head sum -> conv (7 in-ch, width 1536, k=3,
// pad 1) -> split k2/v2/q2 -> scores/softmax/p@v -> write bf16 A matrix.
// One 256-thread block per (b,h). LDS use ~59 KB.
// ---------------------------------------------------------------------------
__global__ __launch_bounds__(256) void preprocess_kernel(
    const float* __restrict__ query, const float* __restrict__ key,
    const float* __restrict__ value, const float* __restrict__ conv_w,
    const float* __restrict__ conv_b, unsigned short* __restrict__ A_bf) {
  __shared__ float xk[7][512];   // conv output, segment 0 (k2)
  __shared__ float xv[7][512];   // segment 1 (v2)
  __shared__ float xq[7][512];   // segment 2 (q2)
  __shared__ float sn[7][520];   // neighbor-sum chunk with +/-1 halo (514 used)
  __shared__ float cw[147];      // conv_w (7,7,3) flat
  __shared__ float cb[7];
  __shared__ float part[49][4];
  __shared__ float sc[7][7];
  __shared__ float pr[7][7];

  const int tid = threadIdx.x;
  const int bh  = blockIdx.x;
  const int b   = bh >> 3;
  const int h   = bh & 7;

  if (tid < 147) cw[tid] = conv_w[tid];
  if (tid < 7)   cb[tid] = conv_b[tid];

  // width w in [0,1536): segment t = w>>9 selects {key, value, query}
  for (int c = 0; c < 3; ++c) {
    __syncthreads();  // previous chunk's conv reads of sn done (also cw/cb)
    for (int idx = tid; idx < 7 * 514; idx += 256) {
      int s = idx / 514;
      int j = idx - s * 514;
      int w = c * 512 - 1 + j;  // global width with halo
      float v = 0.f;
      if (w >= 0 && w < 1536) {
        int t = w >> 9;
        int d = w & 511;
        const float* src = (t == 0) ? key : ((t == 1) ? value : query);
        const float* p = src + ((size_t)b * 7 + s) * 4096 + d;
        int h0 = h - 1;
        #pragma unroll
        for (int dh = 0; dh < 3; ++dh) {
          int hh = h0 + dh;
          if (hh >= 0 && hh < 8) v += p[hh * 512];
        }
      }
      sn[s][j] = v;
    }
    __syncthreads();
    float (*xout)[512] = (c == 0) ? xk : ((c == 1) ? xv : xq);
    for (int idx = tid; idx < 7 * 512; idx += 256) {
      int so = idx >> 9;
      int wl = idx & 511;
      float acc = cb[so];
      #pragma unroll
      for (int i = 0; i < 7; ++i) {
        const float* k3 = &cw[so * 21 + i * 3];
        acc += sn[i][wl] * k3[0] + sn[i][wl + 1] * k3[1] + sn[i][wl + 2] * k3[2];
      }
      xout[so][wl] = acc;
    }
  }
  __syncthreads();

  // scores[sq][sk] = (q2 . k2) / sqrt(512): 49 pairs x 4 partial sums
  if (tid < 196) {
    int p  = tid >> 2;
    int l4 = tid & 3;
    int sq = p / 7, sk = p - sq * 7;
    float a = 0.f;
    int d0 = l4 * 128;
    for (int d = d0; d < d0 + 128; ++d) a += xq[sq][d] * xk[sk][d];
    part[p][l4] = a;
  }
  __syncthreads();
  if (tid < 49) {
    int sq = tid / 7, sk = tid - sq * 7;
    sc[sq][sk] = (part[tid][0] + part[tid][1] + part[tid][2] + part[tid][3]) *
                 0.04419417382415922f;  // 1/sqrt(512)
  }
  __syncthreads();
  if (tid < 7) {
    float m = sc[tid][0];
    #pragma unroll
    for (int k = 1; k < 7; ++k) m = fmaxf(m, sc[tid][k]);
    float e[7];
    float s = 0.f;
    #pragma unroll
    for (int k = 0; k < 7; ++k) { e[k] = __expf(sc[tid][k] - m); s += e[k]; }
    float inv = 1.f / s;
    #pragma unroll
    for (int k = 0; k < 7; ++k) pr[tid][k] = e[k] * inv;
  }
  __syncthreads();

  // atn = p @ v2, write to A matrix row (b*7+sq), cols [h*512, h*512+512)
  for (int idx = tid; idx < 7 * 512; idx += 256) {
    int sq = idx >> 9;
    int d  = idx & 511;
    float a = 0.f;
    #pragma unroll
    for (int k = 0; k < 7; ++k) a += pr[sq][k] * xv[k][d];
    A_bf[((size_t)b * 7 + sq) * 4096 + h * 512 + d] = f2bf(a);
  }
}

// ---------------------------------------------------------------------------
// Stage 3: out[M=7168, N=4096] = A[M,K=4096] * W^T + bias,  W is [N,K] bf16.
// Block tile 128x128, 8 waves (2M x 4N), wave tile 64x32 = 4x2 WMMA tiles.
// Double-buffered LDS; tile fill via async global->LDS when available.
// ---------------------------------------------------------------------------
#define LDST 40  // LDS row stride in elements (80B rows: no 4-way bank conflict)

__global__ __launch_bounds__(256) void gemm_kernel(
    const unsigned short* __restrict__ A, const unsigned short* __restrict__ W,
    const float* __restrict__ bias, float* __restrict__ out) {
  __shared__ __align__(16) unsigned short Asub[2][128][LDST];
  __shared__ __align__(16) unsigned short Bsub[2][128][LDST];

  const int tid   = threadIdx.x;
  const int lane  = tid & 31;
  const int wave  = tid >> 5;
  const int waveM = wave & 1;   // 0..1
  const int waveN = wave >> 1;  // 0..3
  const int g     = (lane >> 4) & 1;  // lane half
  const int mrow  = lane & 15;

  const int mBase = blockIdx.y * 128;
  const int nBase = blockIdx.x * 128;

  // cooperative tile load mapping: 256 threads, each moves 32 bytes per tile
  const int ldRow  = tid >> 1;        // 0..127
  const int ldHalf = (tid & 1) * 16;  // element offset 0 or 16

  const unsigned short* aptr = A + (size_t)(mBase + ldRow) * 4096 + ldHalf;
  const unsigned short* bptr = W + (size_t)(nBase + ldRow) * 4096 + ldHalf;

  v8f acc[4][2];
  const v8f vzero = {0.f, 0.f, 0.f, 0.f, 0.f, 0.f, 0.f, 0.f};
  #pragma unroll
  for (int mi = 0; mi < 4; ++mi)
    #pragma unroll
    for (int ni = 0; ni < 2; ++ni) acc[mi][ni] = vzero;

  const int NK = 4096 / 32;

#if USE_ASYNC_LDS
  // ---- prologue: fill buffer 0 asynchronously (ASYNCcnt += 4) ----
  ASYNC_LOAD_B128(aptr + 0, &Asub[0][ldRow][ldHalf]);
  ASYNC_LOAD_B128(aptr + 8, &Asub[0][ldRow][ldHalf + 8]);
  ASYNC_LOAD_B128(bptr + 0, &Bsub[0][ldRow][ldHalf]);
  ASYNC_LOAD_B128(bptr + 8, &Bsub[0][ldRow][ldHalf + 8]);
#endif

  for (int i = 0; i < NK; ++i) {
    const int buf = i & 1;
#if USE_ASYNC_LDS
    __builtin_amdgcn_s_wait_asynccnt(0);  // this wave's stage-i fill done
    __syncthreads();                      // all waves' fills visible
    if (i + 1 < NK) {
      // overlap: DMA next tile into the other buffer while we do WMMAs
      const int kb1 = (i + 1) * 32;
      const int nb  = buf ^ 1;
      ASYNC_LOAD_B128(aptr + kb1,     &Asub[nb][ldRow][ldHalf]);
      ASYNC_LOAD_B128(aptr + kb1 + 8, &Asub[nb][ldRow][ldHalf + 8]);
      ASYNC_LOAD_B128(bptr + kb1,     &Bsub[nb][ldRow][ldHalf]);
      ASYNC_LOAD_B128(bptr + kb1 + 8, &Bsub[nb][ldRow][ldHalf + 8]);
    }
#else
    // ---- synchronous fallback: VGPR-staged tile fill ----
    {
      const int kb = i * 32;
      __syncthreads();
      const uint4* ap = (const uint4*)(aptr + kb);
      const uint4* bp = (const uint4*)(bptr + kb);
      uint4 a0 = ap[0], a1 = ap[1];
      uint4 b0 = bp[0], b1 = bp[1];
      *(uint4*)&Asub[buf][ldRow][ldHalf]     = a0;
      *(uint4*)&Asub[buf][ldRow][ldHalf + 8] = a1;
      *(uint4*)&Bsub[buf][ldRow][ldHalf]     = b0;
      *(uint4*)&Bsub[buf][ldRow][ldHalf + 8] = b1;
      __syncthreads();
    }
#endif

    // A fragment (16x32 bf16): lane m=mrow; K = 16g+{0..15} contiguous
    v16bf af[4];
    #pragma unroll
    for (int mi = 0; mi < 4; ++mi) {
      int r = waveM * 64 + mi * 16 + mrow;
      v8bf lo = *(const v8bf*)&Asub[buf][r][g * 16];
      v8bf hi = *(const v8bf*)&Asub[buf][r][g * 16 + 8];
      af[mi] = __builtin_shufflevector(lo, hi, 0, 1, 2, 3, 4, 5, 6, 7, 8, 9,
                                       10, 11, 12, 13, 14, 15);
    }
    // B fragment (32x16 bf16): lane n=mrow; K = 16g+{0..15} contiguous.
    // W rows are K-contiguous, so this is a straight 32B read.
    v16bf bfrag[2];
    #pragma unroll
    for (int ni = 0; ni < 2; ++ni) {
      int r = waveN * 32 + ni * 16 + mrow;
      bfrag[ni] = *(const v16bf*)&Bsub[buf][r][g * 16];
    }
    #pragma unroll
    for (int mi = 0; mi < 4; ++mi)
      #pragma unroll
      for (int ni = 0; ni < 2; ++ni)
        acc[mi][ni] = __builtin_amdgcn_wmma_f32_16x16x32_bf16(
            false, af[mi], false, bfrag[ni], (short)0, acc[mi][ni], false,
            false);
  }

  // epilogue: C/D layout -> VGPR j: M = base + j + 8g, N = base + mrow
  #pragma unroll
  for (int ni = 0; ni < 2; ++ni) {
    int col = nBase + waveN * 32 + ni * 16 + mrow;
    float bb = bias[col];
    #pragma unroll
    for (int mi = 0; mi < 4; ++mi) {
      int row0 = mBase + waveM * 64 + mi * 16 + g * 8;
      #pragma unroll
      for (int j = 0; j < 8; ++j) {
        out[(size_t)(row0 + j) * 4096 + col] = acc[mi][ni][j] + bb;
      }
    }
  }
}

// ---------------------------------------------------------------------------
extern "C" void kernel_launch(void* const* d_in, const int* in_sizes, int n_in,
                              void* d_out, int out_size, void* d_ws,
                              size_t ws_size, hipStream_t stream) {
  const float* query  = (const float*)d_in[0];
  const float* key    = (const float*)d_in[1];
  const float* value  = (const float*)d_in[2];
  const float* conv_w = (const float*)d_in[3];
  const float* conv_b = (const float*)d_in[4];
  const float* w_out  = (const float*)d_in[5];
  const float* b_out  = (const float*)d_in[6];
  float* out = (float*)d_out;

  unsigned short* A_bf = (unsigned short*)d_ws;          // 7168*4096 u16
  unsigned short* W_bf = A_bf + (size_t)7168 * 4096;     // 4096*4096 u16

  cvt_w_kernel<<<4096, 256, 0, stream>>>(w_out, W_bf, 4096 * 4096);
  preprocess_kernel<<<8192, 256, 0, stream>>>(query, key, value, conv_w,
                                              conv_b, A_bf);
  dim3 grid(32, 56);  // N tiles x M tiles
  gemm_kernel<<<grid, 256, 0, stream>>>(A_bf, W_bf, b_out, out);
}
